// MultiheadAttention_69234872812274
// MI455X (gfx1250) — matrix operand order repaired and checked by reading, not compiled, not run on hardware
//
#include <hip/hip_runtime.h>
#include <hip/hip_bf16.h>
#include <math.h>
#include <stdint.h>

typedef __bf16 bf16_t;
typedef __attribute__((ext_vector_type(16))) __bf16 v16bf;
typedef __attribute__((ext_vector_type(8)))  __bf16 bf16x8;
typedef __attribute__((ext_vector_type(4)))  __bf16 bf16x4;
typedef __attribute__((ext_vector_type(8)))  float  v8f;
typedef unsigned int u32;
typedef __attribute__((ext_vector_type(4))) u32 v4u;
typedef __attribute__((ext_vector_type(8))) int v8i;
typedef __attribute__((ext_vector_type(4))) int v4i;

#define DEVINL __device__ __forceinline__

// Problem dims
#define BB 4
#define NN 2048
#define CC 1024
#define HH 16
#define DD 64
#define SCALE 0.125f   // 64^-0.5

// Padded LDS row stride (elements): 32 data + 8 pad -> 80B rows, conflict-free frag reads
#define AROW 40

union FragU { v16bf v; bf16x8 h[2]; };

DEVINL v8f zero8() {
    v8f z;
#pragma unroll
    for (int i = 0; i < 8; ++i) z[i] = 0.0f;
    return z;
}

// A fragment from a row-major [M][ld] global matrix (16x32 MxK tile at p).
DEVINL v16bf load_frag_a(const bf16_t* p, int ld, int lane) {
    const int m  = lane & 15;
    const int hi = lane >> 4;
    const bf16_t* r = p + (size_t)m * ld + hi * 8;
    FragU u;
    u.h[0] = *(const bf16x8*)(r);
    u.h[1] = *(const bf16x8*)(r + 16);
    return u.v;
}

// B fragment (32x16 KxN) where B's column n is a contiguous row of length K in memory.
DEVINL v16bf load_frag_b(const bf16_t* p, int ld, int lane) {
    const int n  = lane & 15;
    const int hi = lane >> 4;
    const bf16_t* r = p + (size_t)n * ld + hi * 8;
    FragU u;
    u.h[0] = *(const bf16x8*)(r);
    u.h[1] = *(const bf16x8*)(r + 16);
    return u.v;
}

// LDS fragment loads from TDM-staged, padded [rows][AROW] slabs.
DEVINL v16bf lds_frag_a(const bf16_t* base, int row, int lane) {
    const int hi = lane >> 4;
    const bf16_t* r = base + (size_t)(row + (lane & 15)) * AROW + hi * 8;
    FragU u;
    u.h[0] = *(const bf16x8*)(r);
    u.h[1] = *(const bf16x8*)(r + 16);
    return u.v;
}

DEVINL v16bf lds_frag_b(const bf16_t* base, int col, int lane) {
    const int hi = lane >> 4;
    const bf16_t* r = base + (size_t)(col + (lane & 15)) * AROW + hi * 8;
    FragU u;
    u.h[0] = *(const bf16x8*)(r);
    u.h[1] = *(const bf16x8*)(r + 16);
    return u.v;
}

DEVINL v8f wmma_bf16(v16bf a, v16bf b, v8f c) {
    return __builtin_amdgcn_wmma_f32_16x16x32_bf16(false, a, false, b, (short)0, c,
                                                   false, false);
}

// ---------------------------------------------------------------- TDM helper
// 2D tile load: tile_x=32 bf16 per row, tile_y rows, row stride `stride` elements.
// LDS layout: rows at AROW-element (80B) stride via TDM pad (4 DWORDs per 16 DWORDs).
DEVINL void tdm_load_2d(u32 lds_byte_addr, const bf16_t* gptr,
                        u32 tile_y, u32 dim0, u32 dim1, unsigned long long stride) {
    const unsigned long long ga = (unsigned long long)(uintptr_t)gptr;
    v4u g0;
    g0[0] = 1u;                                   // count=1, user descriptor
    g0[1] = lds_byte_addr;                        // LDS byte address
    g0[2] = (u32)(ga & 0xffffffffu);              // global_addr[31:0]
    g0[3] = (u32)((ga >> 32) & 0x01ffffffu) | (2u << 30);  // global_addr[56:32] | type=2

    const u32 s_lo = (u32)(stride & 0xffffffffu);
    const u32 s_hi = (u32)((stride >> 32) & 0xffffu);
    v8i g1;
    g1[0] = (int)((1u << 16)        // data_size = 1 (2 bytes)
                | (1u << 20)        // pad_enable
                | (3u << 22)        // pad_interval: 16 DWORDs (one 64B row)
                | (3u << 25));      // pad_amount: 4 DWORDs (16B)
    g1[1] = (int)((dim0 & 0xffffu) << 16);                      // tensor_dim0[15:0]
    g1[2] = (int)(((dim0 >> 16) & 0xffffu) | ((dim1 & 0xffffu) << 16));
    g1[3] = (int)(((dim1 >> 16) & 0xffffu) | (32u << 16));      // tile_dim0 = 32
    g1[4] = (int)(tile_y & 0xffffu);                            // tile_dim1; tile_dim2=0
    g1[5] = (int)s_lo;                                          // dim0_stride[31:0]
    g1[6] = (int)s_hi;                                          // dim0_stride[47:32]
    g1[7] = 0;
    const v4i z4 = {0, 0, 0, 0};
    const v8i z8 = {0, 0, 0, 0, 0, 0, 0, 0};
    __builtin_amdgcn_tensor_load_to_lds(g0, g1, z4, z4, z8, 0);
}

// ---------------------------------------------------------------- cast kernel
__global__ __launch_bounds__(256) void cast_f32_bf16(const float* __restrict__ src,
                                                     bf16_t* __restrict__ dst, int n) {
    int i = (blockIdx.x * 256 + threadIdx.x) * 4;
    if (i + 3 < n) {
        float4 f = *(const float4*)(src + i);
        bf16x4 o;
        o[0] = (bf16_t)f.x; o[1] = (bf16_t)f.y; o[2] = (bf16_t)f.z; o[3] = (bf16_t)f.w;
        *(bf16x4*)(dst + i) = o;
    }
}

// ------------------------------------------------------------ QKV projection
// Y[m][e] = sum_c X[m][c] * W[e][c];  M=8192, E=3072, K=1024.
// Block 256x128 staged via TDM into LDS (double-buffered); 8 waves of 64x64 tiles.
__global__ __launch_bounds__(256) void qkv_gemm(const bf16_t* __restrict__ X,
                                                const bf16_t* __restrict__ W,
                                                bf16_t* __restrict__ QKV) {
    __shared__ __align__(16) bf16_t As[2][256 * AROW];
    __shared__ __align__(16) bf16_t Bs[2][128 * AROW];
    const int lane  = threadIdx.x & 31;
    const int wave  = threadIdx.x >> 5;
    const int rowBlk = blockIdx.y * 256;
    const int colBlk = blockIdx.x * 128;
    const int waveM  = (wave >> 1) * 64;
    const int waveN  = (wave & 1) * 64;
    const int K = CC;

    v8f acc[4][4];
#pragma unroll
    for (int i = 0; i < 4; ++i)
#pragma unroll
        for (int j = 0; j < 4; ++j) acc[i][j] = zero8();

    if (wave == 0) {
        tdm_load_2d((u32)(uintptr_t)&As[0][0], X + (size_t)rowBlk * K, 256, K, BB * NN, K);
        tdm_load_2d((u32)(uintptr_t)&Bs[0][0], W + (size_t)colBlk * K, 128, K, 3 * CC,  K);
    }

    for (int k0 = 0; k0 < K; k0 += 32) {
        const int cur = (k0 >> 5) & 1;
        if (wave == 0) __builtin_amdgcn_s_wait_tensorcnt(0);
        __syncthreads();
        if (wave == 0 && (k0 + 32) < K) {
            tdm_load_2d((u32)(uintptr_t)&As[cur ^ 1][0],
                        X + (size_t)rowBlk * K + k0 + 32, 256, K, BB * NN, K);
            tdm_load_2d((u32)(uintptr_t)&Bs[cur ^ 1][0],
                        W + (size_t)colBlk * K + k0 + 32, 128, K, 3 * CC,  K);
        }
        v16bf a[4], b[4];
#pragma unroll
        for (int i = 0; i < 4; ++i) a[i] = lds_frag_a(&As[cur][0], waveM + i * 16, lane);
#pragma unroll
        for (int j = 0; j < 4; ++j) b[j] = lds_frag_b(&Bs[cur][0], waveN + j * 16, lane);
#pragma unroll
        for (int i = 0; i < 4; ++i)
#pragma unroll
            for (int j = 0; j < 4; ++j) acc[i][j] = wmma_bf16(a[i], b[j], acc[i][j]);
    }

    const int hi = lane >> 4, ln = lane & 15;
    const int rowBase = rowBlk + waveM, colBase = colBlk + waveN;
#pragma unroll
    for (int j = 0; j < 4; ++j) {
        const int e = colBase + j * 16 + ln;
        const int s = e >> 10, hd = e & 1023;
        const int h = hd >> 6, d = hd & 63;
        const float scl = (s == 0) ? SCALE : 1.0f;
#pragma unroll
        for (int i = 0; i < 4; ++i) {
#pragma unroll
            for (int r = 0; r < 8; ++r) {
                const int m  = rowBase + i * 16 + r + hi * 8;
                const int bb = m >> 11, n = m & (NN - 1);
                const float val = acc[i][j][r] * scl;
                size_t idx;
                if (s == 2)
                    idx = ((size_t)((2 * BB + bb) * HH + h) * DD + d) * NN + n;
                else
                    idx = ((size_t)((s * BB + bb) * HH + h) * NN + n) * DD + d;
                QKV[idx] = (bf16_t)val;
            }
        }
    }
}

// ------------------------------------------------------- flash attention core
// One wave: one (b,h), 32 q rows; K/V fragments reused by both 16-row subtiles.
__global__ __launch_bounds__(256) void attn_kernel(const bf16_t* __restrict__ QKV,
                                                   bf16_t* __restrict__ O) {
    __shared__ __align__(16) bf16_t lds_p[8][16 * 32];
    const int lane = threadIdx.x & 31;
    const int wave = threadIdx.x >> 5;
    const int wid  = blockIdx.x * 8 + wave;   // 0..4095
    const int bh = wid >> 6;
    const int qt = wid & 63;
    const int b = bh >> 4, h = bh & 15;
    const int hi = lane >> 4, ln = lane & 15;

    const bf16_t* qp = QKV + ((size_t)((0 * BB + b) * HH + h) * NN + qt * 32) * DD;
    const bf16_t* kp = QKV + ((size_t)((1 * BB + b) * HH + h) * NN) * DD;   // [m][d]
    const bf16_t* vp = QKV + ((size_t)((2 * BB + b) * HH + h) * DD) * NN;   // [d][n]

    v16bf aq[2][2];
#pragma unroll
    for (int i = 0; i < 2; ++i)
#pragma unroll
        for (int c = 0; c < 2; ++c)
            aq[i][c] = load_frag_a(qp + (size_t)i * 16 * DD + c * 32, DD, lane);

    v8f o[2][4];
#pragma unroll
    for (int i = 0; i < 2; ++i)
#pragma unroll
        for (int j = 0; j < 4; ++j) o[i][j] = zero8();
    float mrow[2][8], lrow[2][8];
#pragma unroll
    for (int i = 0; i < 2; ++i)
#pragma unroll
        for (int r = 0; r < 8; ++r) { mrow[i][r] = -1.0e30f; lrow[i][r] = 0.0f; }

    bf16_t* P = &lds_p[wave][0];

    for (int m0 = 0; m0 < NN; m0 += 32) {
        v16bf bk[2][2];
#pragma unroll
        for (int t = 0; t < 2; ++t)
#pragma unroll
            for (int c = 0; c < 2; ++c)
                bk[t][c] = load_frag_b(kp + (size_t)(m0 + t * 16) * DD + c * 32, DD, lane);
        v16bf bv[4];
#pragma unroll
        for (int j = 0; j < 4; ++j)
            bv[j] = load_frag_b(vp + (size_t)(j * 16) * NN + m0, NN, lane);

#pragma unroll
        for (int i = 0; i < 2; ++i) {
            v8f s[2];
#pragma unroll
            for (int t = 0; t < 2; ++t) {
                v8f z = zero8();
                z = wmma_bf16(aq[i][0], bk[t][0], z);
                z = wmma_bf16(aq[i][1], bk[t][1], z);
                s[t] = z;
            }
#pragma unroll
            for (int r = 0; r < 8; ++r) {
                const float v0 = s[0][r], v1 = s[1][r];
                float tmax = fmaxf(v0, v1);
#pragma unroll
                for (int off = 1; off < 16; off <<= 1)
                    tmax = fmaxf(tmax, __shfl_xor(tmax, off, 32));
                const float mnew = fmaxf(mrow[i][r], tmax);
                const float corr = __expf(mrow[i][r] - mnew);
                mrow[i][r] = mnew;
                const float p0 = __expf(v0 - mnew);
                const float p1 = __expf(v1 - mnew);
                float ps = p0 + p1;
#pragma unroll
                for (int off = 1; off < 16; off <<= 1)
                    ps += __shfl_xor(ps, off, 32);
                lrow[i][r] = lrow[i][r] * corr + ps;
#pragma unroll
                for (int j = 0; j < 4; ++j) o[i][j][r] *= corr;
                const int n = r + hi * 8;
                P[n * 32 + ln]      = (bf16_t)p0;
                P[n * 32 + 16 + ln] = (bf16_t)p1;
            }
            FragU u;
            {
                const bf16_t* rp = &P[ln * 32 + hi * 8];
                u.h[0] = *(const bf16x8*)(rp);
                u.h[1] = *(const bf16x8*)(rp + 16);
            }
#pragma unroll
            for (int j = 0; j < 4; ++j)
                o[i][j] = wmma_bf16(u.v, bv[j], o[i][j]);
        }
    }

#pragma unroll
    for (int i = 0; i < 2; ++i) {
#pragma unroll
        for (int r = 0; r < 8; ++r) {
            const float inv = 1.0f / lrow[i][r];
            const int n = qt * 32 + i * 16 + r + hi * 8;
            const size_t rowoff = ((size_t)b * NN + n) * CC + h * DD;
#pragma unroll
            for (int j = 0; j < 4; ++j)
                O[rowoff + j * 16 + ln] = (bf16_t)(o[i][j][r] * inv);
        }
    }
}

// ------------------------------------------------------------ out projection
// OUT[m][o] = sum_c A[m][c] * W[o][c] + bias[o];  fp32 output. TDM-staged.
__global__ __launch_bounds__(256) void proj_gemm(const bf16_t* __restrict__ A,
                                                 const bf16_t* __restrict__ W,
                                                 const float* __restrict__ bias,
                                                 float* __restrict__ OUT) {
    __shared__ __align__(16) bf16_t As[2][256 * AROW];
    __shared__ __align__(16) bf16_t Bs[2][128 * AROW];
    const int lane  = threadIdx.x & 31;
    const int wave  = threadIdx.x >> 5;
    const int rowBlk = blockIdx.y * 256;
    const int colBlk = blockIdx.x * 128;
    const int waveM  = (wave >> 1) * 64;
    const int waveN  = (wave & 1) * 64;
    const int K = CC;

    v8f acc[4][4];
#pragma unroll
    for (int i = 0; i < 4; ++i)
#pragma unroll
        for (int j = 0; j < 4; ++j) acc[i][j] = zero8();

    if (wave == 0) {
        tdm_load_2d((u32)(uintptr_t)&As[0][0], A + (size_t)rowBlk * K, 256, K, BB * NN, K);
        tdm_load_2d((u32)(uintptr_t)&Bs[0][0], W + (size_t)colBlk * K, 128, K, CC,      K);
    }

    for (int k0 = 0; k0 < K; k0 += 32) {
        const int cur = (k0 >> 5) & 1;
        if (wave == 0) __builtin_amdgcn_s_wait_tensorcnt(0);
        __syncthreads();
        if (wave == 0 && (k0 + 32) < K) {
            tdm_load_2d((u32)(uintptr_t)&As[cur ^ 1][0],
                        A + (size_t)rowBlk * K + k0 + 32, 256, K, BB * NN, K);
            tdm_load_2d((u32)(uintptr_t)&Bs[cur ^ 1][0],
                        W + (size_t)colBlk * K + k0 + 32, 128, K, CC,      K);
        }
        v16bf a[4], b[4];
#pragma unroll
        for (int i = 0; i < 4; ++i) a[i] = lds_frag_a(&As[cur][0], waveM + i * 16, lane);
#pragma unroll
        for (int j = 0; j < 4; ++j) b[j] = lds_frag_b(&Bs[cur][0], waveN + j * 16, lane);
#pragma unroll
        for (int i = 0; i < 4; ++i)
#pragma unroll
            for (int j = 0; j < 4; ++j) acc[i][j] = wmma_bf16(a[i], b[j], acc[i][j]);
    }

    const int hi = lane >> 4, ln = lane & 15;
    const int rowBase = rowBlk + waveM, colBase = colBlk + waveN;
#pragma unroll
    for (int j = 0; j < 4; ++j) {
        const int col = colBase + j * 16 + ln;
        const float bv = bias[col];
#pragma unroll
        for (int i = 0; i < 4; ++i) {
#pragma unroll
            for (int r = 0; r < 8; ++r) {
                const int m = rowBase + i * 16 + r + hi * 8;
                OUT[(size_t)m * CC + col] = acc[i][j][r] + bv;
            }
        }
    }
}

extern "C" void kernel_launch(void* const* d_in, const int* in_sizes, int n_in,
                              void* d_out, int out_size, void* d_ws, size_t ws_size,
                              hipStream_t stream) {
    const float* x      = (const float*)d_in[0];
    const float* w_qkv  = (const float*)d_in[1];
    const float* w_proj = (const float*)d_in[2];
    const float* b_proj = (const float*)d_in[3];
    float* out = (float*)d_out;

    const size_t nx  = (size_t)BB * NN * CC;   // 8,388,608
    const size_t nwq = (size_t)3 * CC * CC;    // 3,145,728
    const size_t nwp = (size_t)CC * CC;        // 1,048,576

    char* ws = (char*)d_ws;
    bf16_t* x_bf    = (bf16_t*)ws;  ws += nx  * 2;
    bf16_t* wqkv_bf = (bf16_t*)ws;  ws += nwq * 2;
    bf16_t* wproj_bf= (bf16_t*)ws;  ws += nwp * 2;
    bf16_t* qkv_bf  = (bf16_t*)ws;  ws += 3 * nx * 2;
    bf16_t* o_bf    = (bf16_t*)ws;  // nx * 2 bytes

    cast_f32_bf16<<<(int)(nx  / 1024), 256, 0, stream>>>(x,      x_bf,    (int)nx);
    cast_f32_bf16<<<(int)(nwq / 1024), 256, 0, stream>>>(w_qkv,  wqkv_bf, (int)nwq);
    cast_f32_bf16<<<(int)(nwp / 1024), 256, 0, stream>>>(w_proj, wproj_bf,(int)nwp);

    qkv_gemm<<<dim3(3 * CC / 128, BB * NN / 256), 256, 0, stream>>>(x_bf, wqkv_bf, qkv_bf);
    attn_kernel<<<BB * HH * (NN / 32) / 8, 256, 0, stream>>>(qkv_bf, o_bf);
    proj_gemm<<<dim3(CC / 128, BB * NN / 256), 256, 0, stream>>>(o_bf, wproj_bf, b_proj, out);
}
